// ParticleGNN_35424890257917
// MI455X (gfx1250) — compile-verified, matrix operand order checked by reading, not run on hardware
//
#include <hip/hip_runtime.h>
#include <hip/hip_bf16.h>
#include <cstdint>
#include <cstddef>

#define GN 50000
#define GE 640000
#define GD 128
#define GDFF 512
#define GL 3

typedef __bf16 v16bf __attribute__((ext_vector_type(16)));
typedef __bf16 v8bf  __attribute__((ext_vector_type(8)));
typedef float  v8f   __attribute__((ext_vector_type(8)));

static __device__ __forceinline__ __bf16 f2bf(float f) {
    unsigned u = __builtin_bit_cast(unsigned, f);
    u += 0x7FFFu + ((u >> 16) & 1u);              // round to nearest even
    unsigned short h = (unsigned short)(u >> 16);
    return __builtin_bit_cast(__bf16, h);
}

static __device__ __forceinline__ float geluf(float v) {
    return 0.5f * v * (1.0f + erff(v * 0.70710678118654752f));
}

static __device__ __forceinline__ v8f vzero8() {
    v8f z;
#pragma unroll
    for (int i = 0; i < 8; i++) z[i] = 0.0f;
    return z;
}

static __device__ __forceinline__ v8f wmma_bf16(v16bf a, v16bf b, v8f c) {
    return __builtin_amdgcn_wmma_f32_16x16x32_bf16(false, a, false, b, (short)0, c,
                                                   false, false);
}

static __device__ __forceinline__ v16bf pack16(v8bf lo, v8bf hi) {
    v16bf a;
#pragma unroll
    for (int i = 0; i < 8; i++) { a[i] = lo[i]; a[8 + i] = hi[i]; }
    return a;
}

// ---------------------------------------------------------------- utilities
__global__ void k_zero_f32(float* __restrict__ p, int n) {
    for (int i = blockIdx.x * blockDim.x + threadIdx.x; i < n; i += gridDim.x * blockDim.x)
        p[i] = 0.0f;
}

__global__ void k_deg(const int* __restrict__ edst, float* __restrict__ deg, int e) {
    int i = blockIdx.x * blockDim.x + threadIdx.x;
    if (i < e) atomicAdd(&deg[edst[i]], 1.0f);
}

__global__ void k_invdeg(const float* __restrict__ deg, float* __restrict__ inv, int n) {
    int i = blockIdx.x * blockDim.x + threadIdx.x;
    if (i < n) inv[i] = 1.0f / fmaxf(deg[i], 1.0f);
}

__global__ void k_gather_x(const int* __restrict__ ntype, const float* __restrict__ tab,
                           float* __restrict__ x, __bf16* __restrict__ xbf, int total) {
    for (int i = blockIdx.x * blockDim.x + threadIdx.x; i < total; i += gridDim.x * blockDim.x) {
        int n = i >> 7, c = i & 127;
        float v = tab[ntype[n] * GD + c];
        x[i] = v;
        xbf[i] = f2bf(v);
    }
}

__global__ void k_cvt_bf16(const float* __restrict__ s, __bf16* __restrict__ d, int n) {
    int i = blockIdx.x * blockDim.x + threadIdx.x;
    if (i < n) d[i] = f2bf(s[i]);
}

// src: [Lc][K][Nc] f32 row-major  ->  dst: [Lc][Nc][K] bf16 (transposed per layer)
__global__ void k_trans_bf16(const float* __restrict__ src, __bf16* __restrict__ dst,
                             int K, int Nc, int total) {
    for (int idx = blockIdx.x * blockDim.x + threadIdx.x; idx < total;
         idx += gridDim.x * blockDim.x) {
        int per = K * Nc;
        int l = idx / per;
        int rem = idx - l * per;
        int k = rem / Nc;
        int c = rem - k * Nc;
        dst[(size_t)l * per + (size_t)c * K + k] = f2bf(src[idx]);
    }
}

// ---------------------------------------------------------------- message MLP
// wave = 16 edges; block = 8 waves = 128 edges; grid = E/128 = 5000 (exact)
// Dynamic LDS (128KB): w1T stage (64KB) | w2T stage (32KB) | h stage (32KB)
__global__ __launch_bounds__(256) void k_msg(
    const __bf16* __restrict__ xbf, const __bf16* __restrict__ ebf,
    const int* __restrict__ esrc, const int* __restrict__ edst,
    const int* __restrict__ etyp, const float* __restrict__ invdeg,
    const __bf16* __restrict__ w1T /*[128][256]*/, const float* __restrict__ b1,
    const __bf16* __restrict__ w2T /*[128][128]*/, const float* __restrict__ b2,
    float* __restrict__ agg) {
    extern __shared__ char smem[];
    __bf16* w1s = (__bf16*)smem;                       // 32768 elems
    __bf16* w2s = (__bf16*)(smem + 65536);             // 16384 elems
    __bf16* hls = (__bf16*)(smem + 98304);             // 8*16*128 elems

    const int tid = threadIdx.x;
    const int lane = tid & 31;
    const int wave = tid >> 5;
    const int ebase = (blockIdx.x * 8 + wave) * 16;
    const int r = lane & 15;
    const int half = lane >> 4;
    const int kb = half * 8;
    __bf16* hl = &hls[wave * 16 * GD];

    // cooperative weight staging: global -> LDS, b128 granules
    {
        const v8bf* s1 = (const v8bf*)w1T;
        v8bf* d1 = (v8bf*)w1s;
        for (int i = tid; i < 4096; i += 256) d1[i] = s1[i];
        const v8bf* s2 = (const v8bf*)w2T;
        v8bf* d2 = (v8bf*)w2s;
        for (int i = tid; i < 2048; i += 256) d2[i] = s2[i];
    }
    __syncthreads();

    const int e_r = ebase + r;
    const int src = esrc[e_r];
    const int et = etyp[e_r];

    // ---- first matmul: [16,256] @ [256,128]
    v8f acc[8];
#pragma unroll
    for (int t = 0; t < 8; t++) acc[t] = vzero8();

#pragma unroll
    for (int kc = 0; kc < 8; kc++) {
        const int k0 = kc * 32 + kb;
        const __bf16* rowp =
            (kc < 4) ? (xbf + (size_t)src * GD) : (ebf + (size_t)et * GD - GD);
        v8bf alo = *(const v8bf*)(rowp + k0);
        v8bf ahi = *(const v8bf*)(rowp + k0 + 16);
        v16bf a = pack16(alo, ahi);
#pragma unroll
        for (int t = 0; t < 8; t++) {
            const __bf16* wp = w1s + (size_t)(t * 16 + r) * 256 + kc * 32 + kb;
            v16bf b = pack16(*(const v8bf*)wp, *(const v8bf*)(wp + 16));
            acc[t] = wmma_bf16(a, b, acc[t]);
        }
    }

    // bias + GELU -> bf16 stage in LDS (row-major [16][128] per wave)
#pragma unroll
    for (int t = 0; t < 8; t++) {
        const int c = t * 16 + r;
        const float bias = b1[c];
#pragma unroll
        for (int j = 0; j < 8; j++) {
            const int row = j + 8 * half;
            hl[row * GD + c] = f2bf(geluf(acc[t][j] + bias));
        }
    }

    // ---- second matmul: [16,128] @ [128,128]
    v8f acc2[8];
#pragma unroll
    for (int t = 0; t < 8; t++) acc2[t] = vzero8();

#pragma unroll
    for (int kc = 0; kc < 4; kc++) {
        const __bf16* ap = hl + r * GD + kc * 32 + kb;
        v16bf a = pack16(*(const v8bf*)ap, *(const v8bf*)(ap + 16));
#pragma unroll
        for (int t = 0; t < 8; t++) {
            const __bf16* wp = w2s + (size_t)(t * 16 + r) * GD + kc * 32 + kb;
            v16bf b = pack16(*(const v8bf*)wp, *(const v8bf*)(wp + 16));
            acc2[t] = wmma_bf16(a, b, acc2[t]);
        }
    }

    // scale by inv_deg[dst] and scatter-add
    int dstn[8];
    float scl[8];
#pragma unroll
    for (int j = 0; j < 8; j++) {
        int row = j + 8 * half;
        int d = edst[ebase + row];
        dstn[j] = d;
        scl[j] = invdeg[d];
    }
#pragma unroll
    for (int t = 0; t < 8; t++) {
        const int c = t * 16 + r;
        const float bias = b2[c];
#pragma unroll
        for (int j = 0; j < 8; j++) {
            float v = (acc2[t][j] + bias) * scl[j];
            atomicAdd(&agg[(size_t)dstn[j] * GD + c], v);
        }
    }
}

// ---------------------------------------------------------------- update MLP + LN
// wave = 16 nodes; block = 4 waves = 64 nodes; dynamic LDS 64KB (16KB/wave)
__global__ __launch_bounds__(128) void k_upd(
    float* __restrict__ x, __bf16* __restrict__ xbf, const float* __restrict__ agg,
    const __bf16* __restrict__ w1T /*[512][256]*/, const float* __restrict__ b1,
    const __bf16* __restrict__ w2T /*[128][512]*/, const float* __restrict__ b2,
    const float* __restrict__ lng, const float* __restrict__ lnb, int nn) {
    extern __shared__ char smem[];
    const int lane = threadIdx.x & 31;
    const int wave = threadIdx.x >> 5;
    const int nbase = (blockIdx.x * 4 + wave) * 16;
    if (nbase >= nn) return;
    __bf16* hb = (__bf16*)(smem + (size_t)wave * 16 * GDFF * 2);

    const int r = lane & 15;
    const int half = lane >> 4;
    const int kb = half * 8;
    const int node = nbase + r;

    // stage A fragments: concat(x_bf16[node], agg[node]) over 8 K-chunks of 32
    v16bf af[8];
#pragma unroll
    for (int kc = 0; kc < 4; kc++) {
        const __bf16* ap = xbf + (size_t)node * GD + kc * 32 + kb;
        af[kc] = pack16(*(const v8bf*)ap, *(const v8bf*)(ap + 16));
    }
#pragma unroll
    for (int kc = 4; kc < 8; kc++) {
        const float* ap = agg + (size_t)node * GD + (kc - 4) * 32;
        v16bf a;
#pragma unroll
        for (int i = 0; i < 8; i++) {
            a[i] = f2bf(ap[kb + i]);
            a[8 + i] = f2bf(ap[16 + kb + i]);
        }
        af[kc] = a;
    }

    // ---- phase 1: [16,256] @ [256,512] -> GELU -> bf16 LDS [16][512]
    for (int nt = 0; nt < 32; nt++) {
        v8f acc = vzero8();
#pragma unroll
        for (int kc = 0; kc < 8; kc++) {
            const __bf16* wp = w1T + (size_t)(nt * 16 + r) * 256 + kc * 32 + kb;
            v16bf b = pack16(*(const v8bf*)wp, *(const v8bf*)(wp + 16));
            acc = wmma_bf16(af[kc], b, acc);
        }
        const int c = nt * 16 + r;
        const float bias = b1[c];
#pragma unroll
        for (int j = 0; j < 8; j++)
            hb[(j + 8 * half) * GDFF + c] = f2bf(geluf(acc[j] + bias));
    }

    // ---- phase 2: [16,512] @ [512,128]
    v8f acc2[8];
#pragma unroll
    for (int t = 0; t < 8; t++) acc2[t] = vzero8();
    for (int kc = 0; kc < 16; kc++) {
        const __bf16* ap = hb + r * GDFF + kc * 32 + kb;
        v16bf a = pack16(*(const v8bf*)ap, *(const v8bf*)(ap + 16));
#pragma unroll
        for (int t = 0; t < 8; t++) {
            const __bf16* wp = w2T + (size_t)(t * 16 + r) * GDFF + kc * 32 + kb;
            v16bf b = pack16(*(const v8bf*)wp, *(const v8bf*)(wp + 16));
            acc2[t] = wmma_bf16(a, b, acc2[t]);
        }
    }

    // stash h2 + b2 into f32 overlay of this wave's LDS region (8KB < 16KB)
    float* res = (float*)hb;
#pragma unroll
    for (int t = 0; t < 8; t++) {
        const int c = t * 16 + r;
        const float bias = b2[c];
#pragma unroll
        for (int j = 0; j < 8; j++) res[(j + 8 * half) * GD + c] = acc2[t][j] + bias;
    }

    // residual + LayerNorm: lane handles row r, 64 columns (half selects which)
    const int myrow = nbase + r;
    const int c0 = half * 64;
    float s = 0.0f, sq = 0.0f;
    for (int c = 0; c < 64; c++) {
        float v = x[(size_t)myrow * GD + c0 + c] + res[r * GD + c0 + c];
        res[r * GD + c0 + c] = v;
        s += v;
        sq += v * v;
    }
    s += __shfl_xor(s, 16, 32);
    sq += __shfl_xor(sq, 16, 32);
    float mu = s * (1.0f / 128.0f);
    float var = sq * (1.0f / 128.0f) - mu * mu;
    float rs = rsqrtf(var + 1e-5f);
    for (int c = 0; c < 64; c++) {
        int cc = c0 + c;
        float v = (res[r * GD + cc] - mu) * rs * lng[cc] + lnb[cc];
        x[(size_t)myrow * GD + cc] = v;
        xbf[(size_t)myrow * GD + cc] = f2bf(v);
    }
}

// ---------------------------------------------------------------- readout
__global__ void k_pool1(const float* __restrict__ x, float* __restrict__ psum,
                        float* __restrict__ pmax) {
    const int c = threadIdx.x;  // 128
    const int b = blockIdx.x;   // 200
    const int rows = GN / 200;  // 250
    const int r0 = b * rows;
    float s = 0.0f, m = -3.4e38f;
    for (int r = 0; r < rows; r++) {
        float v = x[(size_t)(r0 + r) * GD + c];
        s += v;
        m = fmaxf(m, v);
    }
    psum[b * GD + c] = s;
    pmax[b * GD + c] = m;
}

__global__ void k_pool2(const float* __restrict__ psum, const float* __restrict__ pmax,
                        const float* __restrict__ rw, const float* __restrict__ rb,
                        float* __restrict__ out) {
    __shared__ float pooled[2 * GD];
    const int t = threadIdx.x;  // 256
    if (t < GD) {
        float s = 0.0f;
        for (int i = 0; i < 200; i++) s += psum[i * GD + t];
        pooled[t] = s * (1.0f / (float)GN);
    } else {
        int c = t - GD;
        float m = -3.4e38f;
        for (int i = 0; i < 200; i++) m = fmaxf(m, pmax[i * GD + c]);
        pooled[GD + c] = m;
    }
    __syncthreads();
    if (t < GD) {
        float a = rb[t];
        for (int k = 0; k < 2 * GD; k++) a += pooled[k] * rw[k * GD + t];
        out[t] = a;
    }
}

// ---------------------------------------------------------------- launcher
extern "C" void kernel_launch(void* const* d_in, const int* in_sizes, int n_in,
                              void* d_out, int out_size, void* d_ws, size_t ws_size,
                              hipStream_t stream) {
    const int* node_types = (const int*)d_in[0];
    const int* edge_src = (const int*)d_in[1];
    const int* edge_dst = (const int*)d_in[2];
    const int* edge_types = (const int*)d_in[3];
    const float* node_table = (const float*)d_in[4];
    const float* edge_table = (const float*)d_in[5];
    const float* msg_w1 = (const float*)d_in[6];
    const float* msg_b1 = (const float*)d_in[7];
    const float* msg_w2 = (const float*)d_in[8];
    const float* msg_b2 = (const float*)d_in[9];
    const float* upd_w1 = (const float*)d_in[10];
    const float* upd_b1 = (const float*)d_in[11];
    const float* upd_w2 = (const float*)d_in[12];
    const float* upd_b2 = (const float*)d_in[13];
    const float* ln_g = (const float*)d_in[14];
    const float* ln_b = (const float*)d_in[15];
    const float* readout_w = (const float*)d_in[16];
    const float* readout_b = (const float*)d_in[17];

    char* ws = (char*)d_ws;
    size_t off = 0;
    auto carve = [&](size_t bytes) -> void* {
        void* p = ws + off;
        off += (bytes + 255) & ~(size_t)255;
        return p;
    };
    float* x = (float*)carve((size_t)GN * GD * 4);
    __bf16* xbf = (__bf16*)carve((size_t)GN * GD * 2);
    float* agg = (float*)carve((size_t)GN * GD * 4);
    float* deg = (float*)carve((size_t)GN * 4);
    float* invdeg = (float*)carve((size_t)GN * 4);
    __bf16* etabbf = (__bf16*)carve((size_t)3 * GD * 2);
    __bf16* mw1T = (__bf16*)carve((size_t)GL * 256 * GD * 2);
    __bf16* mw2T = (__bf16*)carve((size_t)GL * GD * GD * 2);
    __bf16* uw1T = (__bf16*)carve((size_t)GL * 256 * GDFF * 2);
    __bf16* uw2T = (__bf16*)carve((size_t)GL * GDFF * GD * 2);
    float* psum = (float*)carve((size_t)200 * GD * 4);
    float* pmax = (float*)carve((size_t)200 * GD * 4);
    (void)in_sizes; (void)n_in; (void)out_size; (void)ws_size;

    // degree -> inverse degree
    k_zero_f32<<<(GN + 255) / 256, 256, 0, stream>>>(deg, GN);
    k_deg<<<(GE + 255) / 256, 256, 0, stream>>>(edge_dst, deg, GE);
    k_invdeg<<<(GN + 255) / 256, 256, 0, stream>>>(deg, invdeg, GN);

    // embeddings + bf16 copies + transposed bf16 weights
    k_gather_x<<<2048, 256, 0, stream>>>(node_types, node_table, x, xbf, GN * GD);
    k_cvt_bf16<<<2, 256, 0, stream>>>(edge_table, etabbf, 3 * GD);
    k_trans_bf16<<<384, 256, 0, stream>>>(msg_w1, mw1T, 256, GD, GL * 256 * GD);
    k_trans_bf16<<<192, 256, 0, stream>>>(msg_w2, mw2T, GD, GD, GL * GD * GD);
    k_trans_bf16<<<1536, 256, 0, stream>>>(upd_w1, uw1T, 256, GDFF, GL * 256 * GDFF);
    k_trans_bf16<<<768, 256, 0, stream>>>(upd_w2, uw2T, GDFF, GD, GL * GDFF * GD);

    const int msg_lds = 65536 + 32768 + 32768;  // w1s + w2s + h stage = 128KB
    for (int l = 0; l < GL; l++) {
        k_zero_f32<<<4096, 256, 0, stream>>>(agg, GN * GD);
        k_msg<<<GE / 128, 256, msg_lds, stream>>>(
            xbf, etabbf, edge_src, edge_dst, edge_types, invdeg,
            mw1T + (size_t)l * 256 * GD, msg_b1 + (size_t)l * GD,
            mw2T + (size_t)l * GD * GD, msg_b2 + (size_t)l * GD, agg);
        k_upd<<<(GN + 63) / 64, 128, 4 * 16 * GDFF * 2, stream>>>(
            x, xbf, agg, uw1T + (size_t)l * 256 * GDFF, upd_b1 + (size_t)l * GDFF,
            uw2T + (size_t)l * GDFF * GD, upd_b2 + (size_t)l * GD,
            ln_g + (size_t)l * GD, ln_b + (size_t)l * GD, GN);
    }

    k_pool1<<<200, 128, 0, stream>>>(x, psum, pmax);
    k_pool2<<<1, 256, 0, stream>>>(psum, pmax, readout_w, readout_b, (float*)d_out);
}
